// TinyTransformerBlock_72610717106728
// MI455X (gfx1250) — compile-verified, matrix operand order checked by reading, not run on hardware
//
#include <hip/hip_runtime.h>
#include <hip/hip_bf16.h>

// ---------------------------------------------------------------------------
// TinyTransformerBlock for MI455X (gfx1250, wave32, WMMA).
// B=4, T=2048, D=1024, H=16, DH=64, FF=4096.  bf16 WMMA, fp32 accumulate.
// Global->LDS staging uses CDNA5 async-to-LDS (ASYNCcnt) with double buffering.
// ---------------------------------------------------------------------------

typedef __attribute__((ext_vector_type(16))) __bf16 bf16x16;
typedef __attribute__((ext_vector_type(8)))  __bf16 bf16x8;
typedef __attribute__((ext_vector_type(4)))  __bf16 bf16x4;
typedef __attribute__((ext_vector_type(8)))  float  f32x8;

#define DEVI static __device__ __forceinline__

// Async copy 16 bytes (this lane) from global memory directly into LDS.
// VDST VGPR holds the workgroup-relative LDS byte offset (HW adds LDS_BASE).
DEVI void async16(unsigned int lds_off, const void* g) {
  asm volatile("global_load_async_to_lds_b128 %0, %1, off"
               :: "v"(lds_off), "v"(g)
               : "memory");
}

DEVI void wait_async0() {
#if __has_builtin(__builtin_amdgcn_s_wait_asynccnt)
  __builtin_amdgcn_s_wait_asynccnt(0);
#else
  asm volatile("s_wait_asynccnt 0x0" ::: "memory");
#endif
}

DEVI unsigned int lds_off32(const void* p) {
  // Generic-pointer low 32 bits == LDS offset within the aperture mapping.
  return (unsigned int)(unsigned long long)p;
}

// Load a 16x32 bf16 WMMA fragment (A or B role) from LDS.
// Per ISA layout: lanes<16 hold K={0..7,16..23}, lanes>=16 hold K={8..15,24..31}
// p points at row start (+ kk*32 for deeper K); off1 = (lane>=16)*8.
DEVI bf16x16 lds_frag(const __bf16* p, int off1) {
  bf16x8 lo = *(const bf16x8*)(p + off1);
  bf16x8 hi = *(const bf16x8*)(p + 16 + off1);
  return __builtin_shufflevector(lo, hi, 0,1,2,3,4,5,6,7,8,9,10,11,12,13,14,15);
}

// Row reductions within 16-lane half-waves (C-matrix rows live per half-wave).
DEVI float red_max16(float x) {
  x = fmaxf(x, __shfl_xor(x, 1, 32));
  x = fmaxf(x, __shfl_xor(x, 2, 32));
  x = fmaxf(x, __shfl_xor(x, 4, 32));
  x = fmaxf(x, __shfl_xor(x, 8, 32));
  return x;
}
DEVI float red_sum16(float x) {
  x += __shfl_xor(x, 1, 32);
  x += __shfl_xor(x, 2, 32);
  x += __shfl_xor(x, 4, 32);
  x += __shfl_xor(x, 8, 32);
  return x;
}

// ---------------------------------------------------------------------------
// fp32 -> bf16 weight conversion (n divisible by 1024)
// ---------------------------------------------------------------------------
__global__ __launch_bounds__(256) void cvt_f32_bf16(const float* __restrict__ in,
                                                    __bf16* __restrict__ out, int n) {
  int i = (blockIdx.x * 256 + threadIdx.x) * 4;
  if (i + 3 < n) {
    float4 v = *(const float4*)(in + i);
    bf16x4 o;
    o[0] = (__bf16)v.x; o[1] = (__bf16)v.y; o[2] = (__bf16)v.z; o[3] = (__bf16)v.w;
    *(bf16x4*)(out + i) = o;
  }
}

// ---------------------------------------------------------------------------
// LayerNorm over D=1024, one block per token, bf16 output (feeds WMMA GEMMs)
// ---------------------------------------------------------------------------
__global__ __launch_bounds__(256) void ln_bf16(const float* __restrict__ x,
                                               const float* __restrict__ g,
                                               const float* __restrict__ be,
                                               __bf16* __restrict__ out) {
  const int row = blockIdx.x;
  const int tid = threadIdx.x;
  const float4 xv = ((const float4*)(x + (size_t)row * 1024))[tid];
  float s = xv.x + xv.y + xv.z + xv.w;
  float q = xv.x * xv.x + xv.y * xv.y + xv.z * xv.z + xv.w * xv.w;
#pragma unroll
  for (int m = 1; m < 32; m <<= 1) {
    s += __shfl_xor(s, m, 32);
    q += __shfl_xor(q, m, 32);
  }
  __shared__ float rs[8], rq[8];
  if ((tid & 31) == 0) { rs[tid >> 5] = s; rq[tid >> 5] = q; }
  __syncthreads();
  float ts = 0.f, tq = 0.f;
#pragma unroll
  for (int i = 0; i < 8; ++i) { ts += rs[i]; tq += rq[i]; }
  const float mean = ts * (1.f / 1024.f);
  const float var  = tq * (1.f / 1024.f) - mean * mean;
  const float rstd = rsqrtf(var + 1e-5f);
  const int c = tid * 4;
  bf16x4 o;
  o[0] = (__bf16)((xv.x - mean) * rstd * g[c + 0] + be[c + 0]);
  o[1] = (__bf16)((xv.y - mean) * rstd * g[c + 1] + be[c + 1]);
  o[2] = (__bf16)((xv.z - mean) * rstd * g[c + 2] + be[c + 2]);
  o[3] = (__bf16)((xv.w - mean) * rstd * g[c + 3] + be[c + 3]);
  ((bf16x4*)(out + (size_t)row * 1024))[tid] = o;
}

// ---------------------------------------------------------------------------
// Tiled bf16 WMMA GEMM:  out[M,N] = A[M,K] @ Bw[N,K]^T + bias (+gelu) (+res)
// Block tile 128x128x32, 256 threads = 8 waves, wave tile 64x32 (4x2 wmma).
// Double-buffered async global->LDS staging: issue tile k+1, compute tile k.
// ---------------------------------------------------------------------------
template <bool GELU, bool RES, bool OUTBF>
__global__ __launch_bounds__(256) void gemm_bf16_wmma(
    const __bf16* __restrict__ A, const __bf16* __restrict__ Bw,
    const float* __restrict__ bias, const float* __restrict__ res,
    float* __restrict__ outf, __bf16* __restrict__ outb,
    int M, int N, int K) {
  constexpr int LDA = 48;  // padded row (bf16) -> keeps 16B alignment, spreads banks
  __shared__ __align__(16) __bf16 As[2][128 * LDA];
  __shared__ __align__(16) __bf16 Bs[2][128 * LDA];

  const int t = threadIdx.x;
  const int lane = t & 31;
  const int ln = lane & 15;
  const int off1 = ((lane >> 4) & 1) * 8;
  const int wid = t >> 5;
  const int wm = wid & 1;   // 0..1 -> 64-row strip
  const int wn = wid >> 1;  // 0..3 -> 32-col strip
  const int bm = blockIdx.x * 128;
  const int bn = blockIdx.y * 128;

  f32x8 acc[4][2];
#pragma unroll
  for (int mi = 0; mi < 4; ++mi)
#pragma unroll
    for (int ni = 0; ni < 2; ++ni)
#pragma unroll
      for (int e = 0; e < 8; ++e) acc[mi][ni][e] = 0.f;

  const int r = t >> 1;
  const int half = t & 1;
  const __bf16* gA = A + (size_t)(bm + r) * K + half * 16;  // + k0
  const __bf16* gB = Bw + (size_t)(bn + r) * K + half * 16;
  const int ldsIdx = r * LDA + half * 16;

  const int KT = K / 32;
  // Prologue: stage tile 0 into buffer 0 (4x async b128 per thread = 64B).
  {
    unsigned int la = lds_off32(&As[0][ldsIdx]);
    unsigned int lb = lds_off32(&Bs[0][ldsIdx]);
    async16(la, gA);      async16(la + 16, gA + 8);
    async16(lb, gB);      async16(lb + 16, gB + 8);
  }

  for (int kt = 0; kt < KT; ++kt) {
    wait_async0();     // staging for this tile complete (this wave)
    __syncthreads();   // ... and for all waves
    if (kt + 1 < KT) {
      const int nb = (kt + 1) & 1;
      const int k1 = (kt + 1) * 32;
      unsigned int la = lds_off32(&As[nb][ldsIdx]);
      unsigned int lb = lds_off32(&Bs[nb][ldsIdx]);
      async16(la, gA + k1);  async16(la + 16, gA + k1 + 8);
      async16(lb, gB + k1);  async16(lb + 16, gB + k1 + 8);
    }

    const int cb = kt & 1;
    bf16x16 aF[4], bF[2];
#pragma unroll
    for (int mi = 0; mi < 4; ++mi)
      aF[mi] = lds_frag(&As[cb][(wm * 64 + mi * 16 + ln) * LDA], off1);
#pragma unroll
    for (int ni = 0; ni < 2; ++ni)
      bF[ni] = lds_frag(&Bs[cb][(wn * 32 + ni * 16 + ln) * LDA], off1);

#pragma unroll
    for (int mi = 0; mi < 4; ++mi)
#pragma unroll
      for (int ni = 0; ni < 2; ++ni)
        acc[mi][ni] = __builtin_amdgcn_wmma_f32_16x16x32_bf16(
            false, aF[mi], false, bF[ni], (short)0, acc[mi][ni], false, false);
    __syncthreads();   // frag reads done before buffer cb is restaged next iter
  }

  // Epilogue: C layout -> lane ln = column, VGPR e -> row e + 8*(lane>=16)
  const int rhi = ((lane >> 4) & 1) * 8;
#pragma unroll
  for (int mi = 0; mi < 4; ++mi) {
#pragma unroll
    for (int ni = 0; ni < 2; ++ni) {
      const int gc = bn + wn * 32 + ni * 16 + ln;
      const float bv = bias[gc];
#pragma unroll
      for (int e = 0; e < 8; ++e) {
        const int gr = bm + wm * 64 + mi * 16 + e + rhi;
        float val = acc[mi][ni][e] + bv;
        if (GELU) val = 0.5f * val * (1.f + erff(val * 0.70710678118654752f));
        if (RES) val += res[(size_t)gr * N + gc];
        if (OUTBF) outb[(size_t)gr * N + gc] = (__bf16)val;
        else       outf[(size_t)gr * N + gc] = val;
      }
    }
  }
}

// ---------------------------------------------------------------------------
// Flash-style non-causal attention. Grid: (B*H, T/64). 128 threads = 4 waves,
// each wave owns a 16-query strip. Q,K,V,out in [B*T, H*DH] bf16.
// Q and K tiles staged with async-to-LDS; V transposed via explicit ds stores.
// ---------------------------------------------------------------------------
__global__ __launch_bounds__(128) void attn_wmma(const __bf16* __restrict__ qg,
                                                 const __bf16* __restrict__ kg,
                                                 const __bf16* __restrict__ vg,
                                                 __bf16* __restrict__ og) {
  constexpr int T = 2048, Dm = 1024, LDS = 80;
  const int bh = blockIdx.x;
  const int b = bh >> 4;
  const int hh = bh & 15;
  const int q0 = blockIdx.y * 64;

  __shared__ __align__(16) __bf16 Qs[64 * LDS];
  __shared__ __align__(16) __bf16 Ks[64 * LDS];
  __shared__ __align__(16) __bf16 Vt[64 * LDS];  // transposed: [dh][key]
  __shared__ __align__(16) __bf16 Ps[64 * LDS];  // softmax probs, per-wave strips

  const int t = threadIdx.x;
  const int lane = t & 31;
  const int ln = lane & 15;
  const int hi8 = (lane >> 4) & 1;
  const int off1 = hi8 * 8;
  const int w = t >> 5;  // wave 0..3
  const int r = t >> 1;
  const int half = t & 1;

  // Stage Q tile (64x64) via async-to-LDS: 4x b128 per thread (64B).
  {
    const __bf16* src = qg + (size_t)(b * T + q0 + r) * Dm + hh * 64 + half * 32;
    unsigned int l0 = lds_off32(&Qs[r * LDS + half * 32]);
#pragma unroll
    for (int j = 0; j < 4; ++j) async16(l0 + 16 * j, src + 8 * j);
  }
  wait_async0();
  __syncthreads();
  bf16x16 aQ[2];
  aQ[0] = lds_frag(&Qs[(w * 16 + ln) * LDS + 0], off1);
  aQ[1] = lds_frag(&Qs[(w * 16 + ln) * LDS + 32], off1);

  f32x8 accO[4];
  float mrun[8], lrun[8];
#pragma unroll
  for (int nd = 0; nd < 4; ++nd)
#pragma unroll
    for (int e = 0; e < 8; ++e) accO[nd][e] = 0.f;
#pragma unroll
  for (int e = 0; e < 8; ++e) { mrun[e] = -INFINITY; lrun[e] = 0.f; }

  for (int kt = 0; kt < T / 64; ++kt) {
    __syncthreads();
    // K tile (row-major: [key][dh]) via async -> contiguous B-frags for Q K^T
    {
      const __bf16* src = kg + (size_t)(b * T + kt * 64 + r) * Dm + hh * 64 + half * 32;
      unsigned int l0 = lds_off32(&Ks[r * LDS + half * 32]);
#pragma unroll
      for (int j = 0; j < 4; ++j) async16(l0 + 16 * j, src + 8 * j);
    }
    // V tile transposed into Vt[dh][key] -> contiguous B-frags for O += P V
    {
      const bf16x8* s8 = (const bf16x8*)(vg + (size_t)(b * T + kt * 64 + r) * Dm + hh * 64 + half * 32);
#pragma unroll
      for (int j = 0; j < 4; ++j) {
        bf16x8 vv = s8[j];
#pragma unroll
        for (int e = 0; e < 8; ++e)
          Vt[(half * 32 + j * 8 + e) * LDS + r] = vv[e];
      }
    }
    wait_async0();
    __syncthreads();

    // S = Q K^T / sqrt(DH): 4 key-subtiles x 2 k-chunks of 32
    f32x8 s[4];
#pragma unroll
    for (int ni = 0; ni < 4; ++ni) {
#pragma unroll
      for (int e = 0; e < 8; ++e) s[ni][e] = 0.f;
#pragma unroll
      for (int kk = 0; kk < 2; ++kk) {
        bf16x16 bK = lds_frag(&Ks[(ni * 16 + ln) * LDS + kk * 32], off1);
        s[ni] = __builtin_amdgcn_wmma_f32_16x16x32_bf16(
            false, aQ[kk], false, bK, (short)0, s[ni], false, false);
      }
#pragma unroll
      for (int e = 0; e < 8; ++e) s[ni][e] *= 0.125f;  // 1/sqrt(64)
    }

    // Online softmax per query row (row e + 8*hi8 of this wave's strip)
#pragma unroll
    for (int e = 0; e < 8; ++e) {
      float rmax = s[0][e];
#pragma unroll
      for (int ni = 1; ni < 4; ++ni) rmax = fmaxf(rmax, s[ni][e]);
      rmax = red_max16(rmax);
      const float mnew = fmaxf(mrun[e], rmax);
      const float corr = __expf(mrun[e] - mnew);
      const int row = w * 16 + e + 8 * hi8;
      float psum = 0.f;
#pragma unroll
      for (int ni = 0; ni < 4; ++ni) {
        const float p = __expf(s[ni][e] - mnew);
        Ps[row * LDS + ni * 16 + ln] = (__bf16)p;
        psum += p;
      }
      psum = red_sum16(psum);
      lrun[e] = lrun[e] * corr + psum;
      mrun[e] = mnew;
#pragma unroll
      for (int nd = 0; nd < 4; ++nd) accO[nd][e] *= corr;
    }
    __syncthreads();  // P visible across lanes before fragment reload

    // O += P V
#pragma unroll
    for (int kk = 0; kk < 2; ++kk) {
      bf16x16 aP = lds_frag(&Ps[(w * 16 + ln) * LDS + kk * 32], off1);
#pragma unroll
      for (int nd = 0; nd < 4; ++nd) {
        bf16x16 bV = lds_frag(&Vt[(nd * 16 + ln) * LDS + kk * 32], off1);
        accO[nd] = __builtin_amdgcn_wmma_f32_16x16x32_bf16(
            false, aP, false, bV, (short)0, accO[nd], false, false);
      }
    }
  }

  // Normalize and write out (bf16)
#pragma unroll
  for (int nd = 0; nd < 4; ++nd) {
    const int col = hh * 64 + nd * 16 + ln;
#pragma unroll
    for (int e = 0; e < 8; ++e) {
      const int row = b * T + q0 + w * 16 + e + 8 * hi8;
      og[(size_t)row * Dm + col] = (__bf16)(accO[nd][e] / lrun[e]);
    }
  }
}

// ---------------------------------------------------------------------------
// Host orchestration
// ---------------------------------------------------------------------------
extern "C" void kernel_launch(void* const* d_in, const int* in_sizes, int n_in,
                              void* d_out, int out_size, void* d_ws, size_t ws_size,
                              hipStream_t stream) {
  (void)in_sizes; (void)n_in; (void)out_size; (void)ws_size;
  constexpr int Bb = 4, T = 2048, D = 1024, FF = 4096;
  constexpr int M = Bb * T;  // 8192 tokens
  const size_t MB = 1024ull * 1024ull;

  const float* x     = (const float*)d_in[0];
  const float* ln1_g = (const float*)d_in[1];
  const float* ln1_b = (const float*)d_in[2];
  const float* ln2_g = (const float*)d_in[3];
  const float* ln2_b = (const float*)d_in[4];
  const float* wq = (const float*)d_in[5];  const float* bq = (const float*)d_in[6];
  const float* wk = (const float*)d_in[7];  const float* bk = (const float*)d_in[8];
  const float* wv = (const float*)d_in[9];  const float* bv = (const float*)d_in[10];
  const float* wo = (const float*)d_in[11]; const float* bo = (const float*)d_in[12];
  const float* w1 = (const float*)d_in[13]; const float* b1 = (const float*)d_in[14];
  const float* w2 = (const float*)d_in[15]; const float* b2 = (const float*)d_in[16];
  float* out = (float*)d_out;

  // Workspace layout (136 MB): ff reuses dead q/k/v + attn span.
  char* ws = (char*)d_ws;
  __bf16* wq_bf = (__bf16*)(ws + 0 * MB);
  __bf16* wk_bf = (__bf16*)(ws + 2 * MB);
  __bf16* wv_bf = (__bf16*)(ws + 4 * MB);
  __bf16* wo_bf = (__bf16*)(ws + 6 * MB);
  __bf16* w1_bf = (__bf16*)(ws + 8 * MB);
  __bf16* w2_bf = (__bf16*)(ws + 16 * MB);
  float*  x2    = (float*) (ws + 24 * MB);   // 32 MB
  __bf16* hbuf  = (__bf16*)(ws + 56 * MB);   // 16 MB (LN1 out, later LN2 out)
  __bf16* qb    = (__bf16*)(ws + 72 * MB);   // 16 MB
  __bf16* kb    = (__bf16*)(ws + 88 * MB);   // 16 MB
  __bf16* vb    = (__bf16*)(ws + 104 * MB);  // 16 MB
  __bf16* attb  = (__bf16*)(ws + 120 * MB);  // 16 MB
  __bf16* ffb   = (__bf16*)(ws + 72 * MB);   // 64 MB, reuses q/k/v/attn after use

  // 1) Weight conversion fp32 -> bf16 (L2-resident afterwards: 24 MB << 192 MB)
  cvt_f32_bf16<<<(D * D) / 1024, 256, 0, stream>>>(wq, wq_bf, D * D);
  cvt_f32_bf16<<<(D * D) / 1024, 256, 0, stream>>>(wk, wk_bf, D * D);
  cvt_f32_bf16<<<(D * D) / 1024, 256, 0, stream>>>(wv, wv_bf, D * D);
  cvt_f32_bf16<<<(D * D) / 1024, 256, 0, stream>>>(wo, wo_bf, D * D);
  cvt_f32_bf16<<<(FF * D) / 1024, 256, 0, stream>>>(w1, w1_bf, FF * D);
  cvt_f32_bf16<<<(D * FF) / 1024, 256, 0, stream>>>(w2, w2_bf, D * FF);

  // 2) LN1 -> h (bf16)
  ln_bf16<<<M, 256, 0, stream>>>(x, ln1_g, ln1_b, hbuf);

  // 3) Q/K/V projections (bias fused, bf16 out)
  dim3 gQKV(M / 128, D / 128);
  gemm_bf16_wmma<false, false, true><<<gQKV, 256, 0, stream>>>(
      hbuf, wq_bf, bq, nullptr, nullptr, qb, M, D, D);
  gemm_bf16_wmma<false, false, true><<<gQKV, 256, 0, stream>>>(
      hbuf, wk_bf, bk, nullptr, nullptr, kb, M, D, D);
  gemm_bf16_wmma<false, false, true><<<gQKV, 256, 0, stream>>>(
      hbuf, wv_bf, bv, nullptr, nullptr, vb, M, D, D);

  // 4) Attention (flash-style, non-causal)
  attn_wmma<<<dim3(Bb * 16, T / 64), 128, 0, stream>>>(qb, kb, vb, attb);

  // 5) Output projection + residual(x) -> x2 (fp32)
  gemm_bf16_wmma<false, true, false><<<dim3(M / 128, D / 128), 256, 0, stream>>>(
      attb, wo_bf, bo, x, x2, nullptr, M, D, D);

  // 6) LN2 -> h2 (bf16, reuses hbuf)
  ln_bf16<<<M, 256, 0, stream>>>(x2, ln2_g, ln2_b, hbuf);

  // 7) FFN up + exact GELU -> ff (bf16)
  gemm_bf16_wmma<true, false, true><<<dim3(M / 128, FF / 128), 256, 0, stream>>>(
      hbuf, w1_bf, b1, nullptr, nullptr, ffb, M, FF, D);

  // 8) FFN down + residual(x2) -> out (fp32)
  gemm_bf16_wmma<false, true, false><<<dim3(M / 128, D / 128), 256, 0, stream>>>(
      ffb, w2_bf, b2, x2, out, nullptr, M, D, FF);
}
